// AtlasTemporalMemoryAttnLayer_31911607009490
// MI455X (gfx1250) — matrix operand order repaired and checked by reading, not compiled, not run on hardware
//
#include <hip/hip_runtime.h>
#include <hip/hip_bf16.h>

// ---------------------------------------------------------------------------
// MI455X (gfx1250) implementation of AtlasTemporalMemoryAttnLayer.
// All GEMMs run on v_wmma_f32_16x16x32_bf16 (fp32 accumulate).
// Weights are pre-packed (bf16, WMMA-B-fragment order) into d_ws; the main
// kernel tiles 16 events per workgroup.  The dominant KV GEMM reuses each B
// fragment across 4 M-tiles held in registers, and the W_mem fragments are
// register-resident for the whole kernel.
// ---------------------------------------------------------------------------

typedef __attribute__((ext_vector_type(16))) __bf16 v16bf;
typedef __attribute__((ext_vector_type(8)))  float  v8f;

// LDS row strides (elements), padded to break bank conflicts (row stride % 64
// dwords == 4 for the bf16 buffers -> 16 fragment rows hit distinct banks).
#define LDA   136   // gathered memory rows, bf16      (64 x 136)
#define LDN   392   // neighbor GEMM input, bf16       (64 x 392)
#define LDKV  260   // KV output / scratch, f32        (64 x 260)
#define LDF   132   // df / Q / attn-out, f32          (16 x 132)
#define LDQI  264   // Q-input / concat input, bf16    (16 x 264)

// Dynamic-LDS byte offsets (all 16B aligned).
#define OFF_A     0u        // 64*136*2 = 17408
#define OFF_NBR   17408u    // 64*392*2 = 50176
#define OFF_KV    67584u    // 64*260*4 = 66560
#define OFF_DF    134144u   // 16*132*4 = 8448
#define OFF_Q     142592u   // 8448
#define OFF_OUT   151040u   // 8448
#define OFF_QIN   159488u   // 16*264*2 = 8448
#define OFF_ATTN  167936u   // 4*2*16*4 = 512
#define SMEM_BYTES 168448u  // ~164.5 KB of the 320 KB WGP LDS

union FragU { v16bf v; uint4 q[2]; };

// A fragment (16x32 bf16 tile) from LDS, ISA 16-bit A layout:
// lanes 0-15: M=lane, K {0..7} in VGPR0-3, K {16..23} in VGPR4-7;
// lanes 16-31: same rows, K offset +8.  -> two ds_load_b128 per lane.
__device__ __forceinline__ v16bf load_a_frag(const __bf16* base, int ld,
                                             int r0, int k0, int lane) {
  int m = lane & 15, half = lane >> 4;
  const __bf16* p = base + (r0 + m) * ld + k0 + half * 8;
  FragU f;
  f.q[0] = *reinterpret_cast<const uint4*>(p);
  f.q[1] = *reinterpret_cast<const uint4*>(p + 16);
  return f.v;
}

// B fragment (32x16 bf16 tile) from pre-swizzled global weights:
// packed[(ntile*ktiles + ktile)*32 + lane][16 bf16] -> two global_load_b128.
__device__ __forceinline__ v16bf load_b_frag(const __bf16* __restrict__ packed,
                                             int ktiles, int ntile, int ktile,
                                             int lane) {
  const __bf16* p = packed + (((size_t)ntile * ktiles + ktile) * 32 + lane) * 16;
  FragU f;
  f.q[0] = *reinterpret_cast<const uint4*>(p);
  f.q[1] = *reinterpret_cast<const uint4*>(p + 8);
  return f.v;
}

// C/D f32 layout: lane = col (0..15), half selects rows 0-7 / 8-15, VGPR j = row j.
__device__ __forceinline__ void store_acc(float* base, int ld, int r0, int n0,
                                          int lane, v8f acc) {
  int col = lane & 15, half = lane >> 4;
  float* p = base + (r0 + half * 8) * ld + n0 + col;
#pragma unroll
  for (int j = 0; j < 8; ++j) p[j * ld] = acc[j];
}

__device__ __forceinline__ v8f wmma_bf16(v16bf a, v16bf b, v8f c) {
  return __builtin_amdgcn_wmma_f32_16x16x32_bf16(false, a, false, b,
                                                 (short)0, c, false, false);
}

// Packed 4x bf16 store (one 8-byte LDS write).
__device__ __forceinline__ void store_bf16x4(__bf16* dst, float a, float b,
                                             float c, float d) {
  union { __bf16 h[4]; uint2 u; } pk;
  pk.h[0] = (__bf16)a; pk.h[1] = (__bf16)b; pk.h[2] = (__bf16)c; pk.h[3] = (__bf16)d;
  *reinterpret_cast<uint2*>(dst) = pk.u;
}

#define ZERO8F {0.f, 0.f, 0.f, 0.f, 0.f, 0.f, 0.f, 0.f}

// ---------------------------------------------------------------------------
// Weight pre-pack: W is [N][Kreal] fp32 row-major (math: x @ W.T, so
// B(k,n) = W[n][k]).  Output: per (ntile,ktile,lane) 16 bf16 in WMMA-B order:
// element j  <->  k = ktile*32 + 8*(lane>>4) + j + (j&8),  n = ntile*16 + (lane&15).
// K padded with zeros up to Kpad (multiple of 32).
// ---------------------------------------------------------------------------
__global__ void pack_weight(const float* __restrict__ W, __bf16* __restrict__ dst,
                            int Kreal, int Kpad, int ntiles) {
  int ktiles = Kpad >> 5;
  int idx = blockIdx.x * blockDim.x + threadIdx.x;
  int total = ntiles * ktiles * 32;
  if (idx >= total) return;
  int lane  = idx & 31;
  int tile  = idx >> 5;
  int ktile = tile % ktiles;
  int ntile = tile / ktiles;
  int n = ntile * 16 + (lane & 15);
  int kbase = ktile * 32 + (lane >> 4) * 8;
  __bf16* d = dst + (size_t)idx * 16;
#pragma unroll
  for (int j = 0; j < 16; ++j) {
    int k = kbase + j + (j & 8);
    float v = (k < Kreal) ? W[(size_t)n * Kreal + k] : 0.f;
    d[j] = (__bf16)v;
  }
}

// ---------------------------------------------------------------------------
// Main kernel: one workgroup = 16 events (256 neighbor rows, chunked 4x64).
// ---------------------------------------------------------------------------
__global__ __launch_bounds__(256)
void tga_main(const float* __restrict__ memory, const float* __restrict__ dst_feat,
              const float* __restrict__ src_feat, const float* __restrict__ edge_feat,
              const float* __restrict__ dst_ts, const float* __restrict__ src_ts,
              const int* __restrict__ dst_nodes, const int* __restrict__ src_nodes,
              const float* __restrict__ b_mem, const float* __restrict__ time_w,
              const float* __restrict__ time_b, const float* __restrict__ b_q,
              const float* __restrict__ b_kv, const float* __restrict__ b_out,
              const float* __restrict__ ln_g, const float* __restrict__ ln_b,
              const __bf16* __restrict__ Wmem_p, const __bf16* __restrict__ Wq_p,
              const __bf16* __restrict__ Wkv_p, const __bf16* __restrict__ Wout_p,
              float* __restrict__ out) {
  extern __shared__ char smem[];
  __bf16* sA    = (__bf16*)(smem + OFF_A);     // gathered memory rows (bf16)
  __bf16* sNbr  = (__bf16*)(smem + OFF_NBR);   // [sf | edge | t_src] bf16
  float*  sKV   = (float*) (smem + OFF_KV);    // KV chunk f32 (also GEMM scratch)
  float*  sDf   = (float*) (smem + OFF_DF);    // df f32
  float*  sQ    = (float*) (smem + OFF_Q);     // Q f32
  float*  sOut  = (float*) (smem + OFF_OUT);   // attn out -> final out, f32
  __bf16* sQin  = (__bf16*)(smem + OFF_QIN);   // Q-GEMM input / concat input
  float*  sAttn = (float*) (smem + OFF_ATTN);  // softmax weights [4][2][16]

  const int tid  = threadIdx.x;
  const int wave = tid >> 5;
  const int lane = tid & 31;
  const int e0   = blockIdx.x * 16;

  // W_mem B fragments: register-resident for GEMM1a and all GEMM1b chunks.
  v16bf wb_mem[4];
#pragma unroll
  for (int kt = 0; kt < 4; ++kt)
    wb_mem[kt] = load_b_frag(Wmem_p, 4, wave, kt, lane);

  // ---------------- Stage 1: dst path ----------------
  // Gather 16 dst memory rows -> bf16 LDS (float4 loads, L2-resident table).
  for (int i = tid; i < 16 * 32; i += 256) {
    int r = i >> 5, c4 = i & 31;
    int node = dst_nodes[e0 + r];
    float4 v = ((const float4*)(memory + (size_t)node * 128))[c4];
    store_bf16x4(sA + r * LDA + c4 * 4, v.x, v.y, v.z, v.w);
  }
  __syncthreads();

  // GEMM1a: df_raw(16x128) = memRows(16x128) x Wmem(128x128). Wave w -> N-tile w.
  {
    v8f acc = ZERO8F;
#pragma unroll
    for (int kt = 0; kt < 4; ++kt) {
      v16bf a = load_a_frag(sA, LDA, 0, kt * 32, lane);
      acc = wmma_bf16(a, wb_mem[kt], acc);
    }
    store_acc(sDf, LDF, 0, wave * 16, lane, acc);
  }
  __syncthreads();

  // df = raw + dst_feat + b_mem; also fill Q-input [df | cos(t_dst) | 0-pad].
  for (int i = tid; i < 16 * 32; i += 256) {
    int r = i >> 5, c = (i & 31) * 4;
    float4 dfv = *(const float4*)(dst_feat + (size_t)(e0 + r) * 128 + c);
    float4 bm  = *(const float4*)(b_mem + c);
    float* dp = sDf + r * LDF + c;
    float x0 = dp[0] + dfv.x + bm.x, x1 = dp[1] + dfv.y + bm.y;
    float x2 = dp[2] + dfv.z + bm.z, x3 = dp[3] + dfv.w + bm.w;
    dp[0] = x0; dp[1] = x1; dp[2] = x2; dp[3] = x3;
    store_bf16x4(sQin + r * LDQI + c, x0, x1, x2, x3);
  }
  for (int i = tid; i < 16 * 32; i += 256) {
    int r = i >> 5, d = (i & 31) * 4;   // 100 and 128 are multiples of 4
    float x0 = 0.f, x1 = 0.f, x2 = 0.f, x3 = 0.f;
    if (d < 100) {
      float t = dst_ts[e0 + r];
      x0 = __cosf(t * time_w[d]     + time_b[d]);
      x1 = __cosf(t * time_w[d + 1] + time_b[d + 1]);
      x2 = __cosf(t * time_w[d + 2] + time_b[d + 2]);
      x3 = __cosf(t * time_w[d + 3] + time_b[d + 3]);
    }
    store_bf16x4(sQin + r * LDQI + 128 + d, x0, x1, x2, x3);
  }
  __syncthreads();

  // GEMM2: Q(16x128) = q_in(16x256) x Wq(256x128). Wave w -> N-tile w.
  {
    v8f acc = ZERO8F;
#pragma unroll
    for (int kt = 0; kt < 8; ++kt) {
      v16bf a = load_a_frag(sQin, LDQI, 0, kt * 32, lane);
      v16bf b = load_b_frag(Wq_p, 8, wave, kt, lane);
      acc = wmma_bf16(a, b, acc);
    }
    store_acc(sQ, LDF, 0, wave * 16, lane, acc);
  }
  __syncthreads();
  for (int i = tid; i < 16 * 32; i += 256) {
    int r = i >> 5, c = (i & 31) * 4;
    float4 bq = *(const float4*)(b_q + c);
    float* qp = sQ + r * LDF + c;
    qp[0] += bq.x; qp[1] += bq.y; qp[2] += bq.z; qp[3] += bq.w;
  }
  __syncthreads();

  // ---------------- Stage 2: neighbor chunks (64 rows = 4 events each) -------
  for (int chunk = 0; chunk < 4; ++chunk) {
    // Gather 64 src memory rows -> bf16 LDS.
    for (int i = tid; i < 64 * 32; i += 256) {
      int r = i >> 5, c4 = i & 31;
      int el = chunk * 4 + (r >> 4), k = r & 15;
      int node = src_nodes[(e0 + el) * 16 + k];
      float4 v = ((const float4*)(memory + (size_t)node * 128))[c4];
      store_bf16x4(sA + r * LDA + c4 * 4, v.x, v.y, v.z, v.w);
    }
    __syncthreads();

    // GEMM1b: sf_raw(64x128) = memRows(64x128) x Wmem; scratch into sKV.
    // Wave w owns N-tile w for all 4 M-tiles (B frags are register-resident).
    {
      v8f acc[4] = {ZERO8F, ZERO8F, ZERO8F, ZERO8F};
#pragma unroll
      for (int kt = 0; kt < 4; ++kt) {
#pragma unroll
        for (int mt = 0; mt < 4; ++mt) {
          v16bf a = load_a_frag(sA, LDA, mt * 16, kt * 32, lane);
          acc[mt] = wmma_bf16(a, wb_mem[kt], acc[mt]);
        }
      }
#pragma unroll
      for (int mt = 0; mt < 4; ++mt)
        store_acc(sKV, LDKV, mt * 16, wave * 16, lane, acc[mt]);
    }
    __syncthreads();

    // nbr_in[:,0:128] = bf16(sf_raw + src_feat + b_mem)
    for (int i = tid; i < 64 * 32; i += 256) {
      int r = i >> 5, c = (i & 31) * 4;
      int el = chunk * 4 + (r >> 4), k = r & 15;
      float4 sfv = *(const float4*)(src_feat + ((size_t)(e0 + el) * 16 + k) * 128 + c);
      float4 bm  = *(const float4*)(b_mem + c);
      const float* kp = sKV + r * LDKV + c;
      store_bf16x4(sNbr + r * LDN + c,
                   kp[0] + sfv.x + bm.x, kp[1] + sfv.y + bm.y,
                   kp[2] + sfv.z + bm.z, kp[3] + sfv.w + bm.w);
    }
    // nbr_in[:,128:256] = edge_feat ; [:,256:384] = cos(dt) time enc (+0-pad)
    for (int i = tid; i < 64 * 64; i += 256) {
      int r = i >> 6, c = (i & 63) * 4;   // c in [0,256): edge | time
      int el = chunk * 4 + (r >> 4), k = r & 15;
      float x0 = 0.f, x1 = 0.f, x2 = 0.f, x3 = 0.f;
      if (c < 128) {
        float4 ev = *(const float4*)(edge_feat + ((size_t)(e0 + el) * 16 + k) * 128 + c);
        x0 = ev.x; x1 = ev.y; x2 = ev.z; x3 = ev.w;
      } else {
        int d = c - 128;
        if (d < 100) {
          float dt = dst_ts[e0 + el] - src_ts[(e0 + el) * 16 + k];
          dt = fmaxf(dt, 0.f);
          x0 = __cosf(dt * time_w[d]     + time_b[d]);
          x1 = __cosf(dt * time_w[d + 1] + time_b[d + 1]);
          x2 = __cosf(dt * time_w[d + 2] + time_b[d + 2]);
          x3 = __cosf(dt * time_w[d + 3] + time_b[d + 3]);
        }
      }
      store_bf16x4(sNbr + r * LDN + 128 + c, x0, x1, x2, x3);
    }
    __syncthreads();

    // GEMM3 (dominant): KV(64x256) = nbr_in(64x384) x Wkv(384x256).
    // Wave w owns N-tiles {2w, 2w+1} x all 4 M-tiles; each B fragment is
    // reused by 4 WMMAs (4x less B traffic from L2).
    {
      v8f acc[8] = {ZERO8F, ZERO8F, ZERO8F, ZERO8F, ZERO8F, ZERO8F, ZERO8F, ZERO8F};
#pragma unroll
      for (int kt = 0; kt < 12; ++kt) {
        v16bf a0 = load_a_frag(sNbr, LDN,  0, kt * 32, lane);
        v16bf a1 = load_a_frag(sNbr, LDN, 16, kt * 32, lane);
        v16bf a2 = load_a_frag(sNbr, LDN, 32, kt * 32, lane);
        v16bf a3 = load_a_frag(sNbr, LDN, 48, kt * 32, lane);
#pragma unroll
        for (int j = 0; j < 2; ++j) {
          v16bf b = load_b_frag(Wkv_p, 12, wave * 2 + j, kt, lane);
          acc[j * 4 + 0] = wmma_bf16(a0, b, acc[j * 4 + 0]);
          acc[j * 4 + 1] = wmma_bf16(a1, b, acc[j * 4 + 1]);
          acc[j * 4 + 2] = wmma_bf16(a2, b, acc[j * 4 + 2]);
          acc[j * 4 + 3] = wmma_bf16(a3, b, acc[j * 4 + 3]);
        }
      }
#pragma unroll
      for (int j = 0; j < 2; ++j)
#pragma unroll
        for (int mt = 0; mt < 4; ++mt)
          store_acc(sKV, LDKV, mt * 16, (wave * 2 + j) * 16, lane, acc[j * 4 + mt]);
    }
    __syncthreads();
    for (int i = tid; i < 64 * 64; i += 256) {
      int r = i >> 6, c = (i & 63) * 4;
      float4 bk = *(const float4*)(b_kv + c);
      float* kp = sKV + r * LDKV + c;
      kp[0] += bk.x; kp[1] += bk.y; kp[2] += bk.z; kp[3] += bk.w;
    }
    __syncthreads();

    // Attention: waves 0..3, one event each. lane = h*16+k.
    if (wave < 4) {
      int el = chunk * 4 + wave;
      int h = lane >> 4, k = lane & 15;
      int r = wave * 16 + k;
      const float4* qp = (const float4*)(sQ + el * LDF + h * 64);
      const float4* kp = (const float4*)(sKV + r * LDKV + h * 64);
      float s = 0.f;
#pragma unroll
      for (int d = 0; d < 16; ++d) {
        float4 qv = qp[d], kv = kp[d];
        s += qv.x * kv.x + qv.y * kv.y + qv.z * kv.z + qv.w * kv.w;
      }
      s = s > 0.f ? s : 0.2f * s;                        // leaky_relu(0.2)
      float m = s;
      for (int off = 1; off < 16; off <<= 1) m = fmaxf(m, __shfl_xor(m, off, 16));
      float p = __expf(s - m);
      float sum = p;
      for (int off = 1; off < 16; off <<= 1) sum += __shfl_xor(sum, off, 16);
      sAttn[(wave * 2 + h) * 16 + k] = p / sum;
    }
    __syncthreads();

    // out[e] = sum_k attn * V ; V = sKV cols 128..255 (head h = c>>6).
    for (int i = tid; i < 4 * 32; i += 256) {
      int er = i >> 5, c = (i & 31) * 4, h = c >> 6;
      float a0 = 0.f, a1 = 0.f, a2 = 0.f, a3 = 0.f;
#pragma unroll
      for (int k = 0; k < 16; ++k) {
        float w = sAttn[(er * 2 + h) * 16 + k];
        const float* vp = sKV + (er * 16 + k) * LDKV + 128 + c;
        a0 += w * vp[0]; a1 += w * vp[1]; a2 += w * vp[2]; a3 += w * vp[3];
      }
      float* op = sOut + (chunk * 4 + er) * LDF + c;
      op[0] = a0; op[1] = a1; op[2] = a2; op[3] = a3;
    }
    __syncthreads();
  }

  // ---------------- Stage 3: output projection + LayerNorm ----------------
  // concat [attn_out | df] -> bf16 (reuse sQin)
  for (int i = tid; i < 16 * 64; i += 256) {
    int r = i >> 6, c = (i & 63) * 4;
    const float* sp = (c < 128) ? (sOut + r * LDF + c) : (sDf + r * LDF + (c - 128));
    store_bf16x4(sQin + r * LDQI + c, sp[0], sp[1], sp[2], sp[3]);
  }
  __syncthreads();

  // GEMM4: out2(16x128) = cat(16x256) x Wout(256x128). Wave w -> N-tile w.
  {
    v8f acc = ZERO8F;
#pragma unroll
    for (int kt = 0; kt < 8; ++kt) {
      v16bf a = load_a_frag(sQin, LDQI, 0, kt * 32, lane);
      v16bf b = load_b_frag(Wout_p, 8, wave, kt, lane);
      acc = wmma_bf16(a, b, acc);
    }
    store_acc(sOut, LDF, 0, wave * 16, lane, acc);
  }
  __syncthreads();

  // bias + ReLU + LayerNorm; wave w handles rows w and w+8, 4 cols/lane.
#pragma unroll
  for (int rr = 0; rr < 2; ++rr) {
    int r = wave + rr * 8;
    int c = lane * 4;
    float4 xv = *(const float4*)(sOut + r * LDF + c);
    float4 bo = *(const float4*)(b_out + c);
    float v0 = fmaxf(xv.x + bo.x, 0.f), v1 = fmaxf(xv.y + bo.y, 0.f);
    float v2 = fmaxf(xv.z + bo.z, 0.f), v3 = fmaxf(xv.w + bo.w, 0.f);
    float sum = v0 + v1 + v2 + v3;
    float sq  = v0 * v0 + v1 * v1 + v2 * v2 + v3 * v3;
    for (int off = 1; off < 32; off <<= 1) {
      sum += __shfl_xor(sum, off, 32);
      sq  += __shfl_xor(sq,  off, 32);
    }
    float mu   = sum * (1.f / 128.f);
    float var  = sq * (1.f / 128.f) - mu * mu;
    float rstd = rsqrtf(var + 1e-5f);
    float4 g = *(const float4*)(ln_g + c);
    float4 bb = *(const float4*)(ln_b + c);
    float4 ov;
    ov.x = (v0 - mu) * rstd * g.x + bb.x;
    ov.y = (v1 - mu) * rstd * g.y + bb.y;
    ov.z = (v2 - mu) * rstd * g.z + bb.z;
    ov.w = (v3 - mu) * rstd * g.w + bb.w;
    *(float4*)(out + (size_t)(e0 + r) * 128 + c) = ov;
  }
}

// ---------------------------------------------------------------------------
extern "C" void kernel_launch(void* const* d_in, const int* in_sizes, int n_in,
                              void* d_out, int out_size, void* d_ws, size_t ws_size,
                              hipStream_t stream) {
  const float* memory    = (const float*)d_in[0];
  const float* dst_feat  = (const float*)d_in[1];
  const float* src_feat  = (const float*)d_in[2];
  const float* edge_feat = (const float*)d_in[3];
  const float* dst_ts    = (const float*)d_in[4];
  const float* src_ts    = (const float*)d_in[5];
  const int*   dst_nodes = (const int*)d_in[6];
  const int*   src_nodes = (const int*)d_in[7];
  const float* W_mem     = (const float*)d_in[8];
  const float* b_mem     = (const float*)d_in[9];
  const float* time_w    = (const float*)d_in[10];
  const float* time_b    = (const float*)d_in[11];
  const float* W_q       = (const float*)d_in[12];
  const float* b_q       = (const float*)d_in[13];
  const float* W_kv      = (const float*)d_in[14];
  const float* b_kv      = (const float*)d_in[15];
  const float* W_out     = (const float*)d_in[16];
  const float* b_out     = (const float*)d_in[17];
  const float* ln_g      = (const float*)d_in[18];
  const float* ln_b      = (const float*)d_in[19];
  float* outp = (float*)d_out;

  // Workspace layout (bf16 packed weights): 32KB + 64KB + 192KB + 64KB.
  __bf16* Wmem_p = (__bf16*)d_ws;
  __bf16* Wq_p   = Wmem_p + (size_t)8 * 4 * 32 * 16;
  __bf16* Wkv_p  = Wq_p   + (size_t)8 * 8 * 32 * 16;
  __bf16* Wout_p = Wkv_p  + (size_t)16 * 12 * 32 * 16;

  pack_weight<<<dim3((8 * 4 * 32 + 255) / 256),   dim3(256), 0, stream>>>(W_mem, Wmem_p, 128, 128, 8);
  pack_weight<<<dim3((8 * 8 * 32 + 255) / 256),   dim3(256), 0, stream>>>(W_q,   Wq_p,   228, 256, 8);
  pack_weight<<<dim3((16 * 12 * 32 + 255) / 256), dim3(256), 0, stream>>>(W_kv,  Wkv_p,  356, 384, 16);
  pack_weight<<<dim3((8 * 8 * 32 + 255) / 256),   dim3(256), 0, stream>>>(W_out, Wout_p, 256, 256, 8);

  int nL = in_sizes[4];          // 50000, divisible by 16
  int blocks = nL / 16;
  tga_main<<<dim3(blocks), dim3(256), SMEM_BYTES, stream>>>(
      memory, dst_feat, src_feat, edge_feat, dst_ts, src_ts, dst_nodes, src_nodes,
      b_mem, time_w, time_b, b_q, b_kv, b_out, ln_g, ln_b,
      Wmem_p, Wq_p, Wkv_p, Wout_p, outp);
}
